// NeuralODE_41686952575789
// MI455X (gfx1250) — compile-verified, matrix operand order checked by reading, not run on hardware
//
#include <hip/hip_runtime.h>
#include <hip/hip_bf16.h>

// ---------------------------------------------------------------------------
// Latent Neural ODE on MI455X (gfx1250), wave32 + WMMA f32_16x16x32_f16.
// One wave (32 threads) owns a 16-row batch tile and runs the entire 255-step
// RK6 scan with the gen-MLP GEMMs on the matrix engine. All weights and the
// ODE state live in LDS for the whole kernel. Activation epilogues use fast
// trans ops (v_exp/v_log/v_tanh) with no divergence.
// ---------------------------------------------------------------------------

typedef __attribute__((ext_vector_type(16))) _Float16 v16h;
typedef __attribute__((ext_vector_type(8)))  float    v8f;

#define B_TOT 2048
#define T_TOT 256
#define ACT_S 72   // activation row stride (halves), 16B-aligned rows
#define W1_S  40   // gen W1 row stride (halves), K padded 16->32
#define W_S   72   // gen W2/W3/W4 row stride (halves)

// flat output layout: x_recons | x_factors[:, -1] | mu | logvar
#define XREC_OFF 0
#define XFAC_OFF (B_TOT * T_TOT * 2)
#define MU_OFF   (XFAC_OFF + B_TOT * 32)
#define LV_OFF   (MU_OFF + B_TOT * 16)

// RK tableau (from reference)
__constant__ float c_Cs[6] = {0.0f, 0.161f, 0.327f, 0.9f, 0.9800255409045097f, 1.0f};
__constant__ float c_A[6][5] = {
    {0.f, 0.f, 0.f, 0.f, 0.f},
    {0.161f, 0.f, 0.f, 0.f, 0.f},
    {-0.008480655492356989f, 0.335480655492357f, 0.f, 0.f, 0.f},
    {2.8971530571054935f, -6.359448489975075f, 4.3622954328695815f, 0.f, 0.f},
    {5.325864828439257f, -11.748883564062828f, 7.4955393428898365f, -0.09249506636175525f, 0.f},
    {5.86145544294642f, -12.92096931784711f, 8.159367898576159f, -0.071584973281401f, -0.028269050394068383f}};
__constant__ float c_B[6] = {0.09646076681806523f, 0.01f, 0.4798896504144996f,
                             1.379008574103742f, -3.290069515436081f, 2.324710524099774f};

struct Params {
    const float *xs, *te;
    const float *eW0, *eB0, *eW1, *eB1, *eW2, *eB2, *eW3, *eB3;
    const float *gW0, *gB0, *gW1, *gB1, *gW2, *gB2, *gW3, *gB3;
    const float *dfW, *dfb, *drW, *drb;
    float* out;
};

// Branch-free, numerically stable softplus: max(x,0) + log(1 + exp(-|x|)).
// Lowers to v_exp_f32 / v_log_f32 trans ops, no divergence.
__device__ __forceinline__ float softplus_f(float x) {
    const float e = __expf(-fabsf(x));
    return fmaxf(x, 0.f) + __logf(1.f + e);
}

// CDNA5 hardware tanh (TRANS op). v_nop covers the documented TRANS
// co-execution hazard (1 independent op before the result is consumed).
__device__ __forceinline__ float tanh_f(float x) {
    float r;
    asm volatile("v_tanh_f32 %0, %1\n\tv_nop" : "=v"(r) : "v"(x));
    return r;
}

// Wave-local LDS ordering: all staging buffers are private to this wave and
// workgroups are single-wave, so a compiler scheduling barrier + DScnt drain
// is exactly the required synchronization (covers ds_store and ds_add_f32).
__device__ __forceinline__ void wave_sync() {
    __builtin_amdgcn_wave_barrier();
    asm volatile("s_wait_dscnt 0" ::: "memory");
    __builtin_amdgcn_wave_barrier();
}

// Load one 16-lane fragment row: two contiguous 8-half (16B) K-runs.
// CDNA5 16-bit A layout: lane<16 -> M=lane,  K runs {k0..k0+7},{k0+16..k0+23}
//                        lane>=16 -> M=lane-16, K runs {k0+8..15},{k0+24..31}
// (B layout is the mirror with N = lane&15; same loader applies.)
__device__ __forceinline__ v16h ld_frag(const _Float16* rowbase, int kbase, int lane) {
    const _Float16* p = rowbase + kbase + ((lane >> 4) << 3);
    v16h a;
    __builtin_memcpy(&a, p, 16);
    __builtin_memcpy(reinterpret_cast<char*>(&a) + 16, p + 16, 16);
    return a;
}

// One gen-MLP layer: D = softplus(A(16xK) * W^T + b), A from `in`, result -> `out`.
// A-fragments are hoisted out of the N-tile loop (reused by all output tiles).
template <int NT, int KC>
__device__ __forceinline__ void gen_layer(const _Float16* in, const _Float16* w, int wS,
                                          const float* bias, _Float16* out, int lane) {
    const int n = lane & 15;
    const int mbase = (lane >> 4) << 3;
    v16h a[KC];
#pragma unroll
    for (int kc = 0; kc < KC; ++kc) a[kc] = ld_frag(in + n * ACT_S, kc * 32, lane);
#pragma unroll
    for (int nt = 0; nt < NT; ++nt) {
        const float bv = bias[nt * 16 + n];
        v8f c;
#pragma unroll
        for (int e = 0; e < 8; ++e) c[e] = bv;
#pragma unroll
        for (int kc = 0; kc < KC; ++kc) {
            v16h bf = ld_frag(w + (nt * 16 + n) * wS, kc * 32, lane);
            c = __builtin_amdgcn_wmma_f32_16x16x32_f16(false, a[kc], false, bf, (short)0, c,
                                                       false, false);
        }
#pragma unroll
        for (int e = 0; e < 8; ++e)
            out[(mbase + e) * ACT_S + nt * 16 + n] = (_Float16)softplus_f(c[e]);
    }
}

__global__ __launch_bounds__(32) void node_wmma_kernel(Params P) {
    // ---- LDS (~46 KB) ----
    __shared__ __align__(16) _Float16 s_w1[64 * W1_S];
    __shared__ __align__(16) _Float16 s_w2[64 * W_S];
    __shared__ __align__(16) _Float16 s_w3[64 * W_S];
    __shared__ __align__(16) _Float16 s_w4[48 * W_S];
    __shared__ float s_gb1[64], s_gb2[64], s_gb3[64], s_gb4[48];
    __shared__ __align__(16) _Float16 s_actA[16 * ACT_S];
    __shared__ __align__(16) _Float16 s_actB[16 * ACT_S];
    __shared__ float s_y[256];        // y[r][l], 16x16 state
    __shared__ float s_k[6 * 256];    // k1..k6
    __shared__ float s_pm[64], s_d0[64], s_d1[64], s_dx[64];  // [16][4] padded
    __shared__ float s_df[512], s_dfb[32], s_dr[64], s_drb[2];

    const int lane  = threadIdx.x;
    const int bbase = blockIdx.x * 16;

    // ---- stage gen/decoder weights into LDS (f16 for WMMA B operands) ----
    for (int idx = lane; idx < 64 * 32; idx += 32) {      // W1: K 16 -> padded 32
        int nr = idx >> 5, k = idx & 31;
        s_w1[nr * W1_S + k] = (_Float16)((k < 16) ? P.gW0[nr * 16 + k] : 0.f);
    }
    for (int idx = lane; idx < 64 * 64; idx += 32) {
        int nr = idx >> 6, k = idx & 63;
        s_w2[nr * W_S + k] = (_Float16)P.gW1[idx];
        s_w3[nr * W_S + k] = (_Float16)P.gW2[idx];
    }
    for (int idx = lane; idx < 48 * 64; idx += 32) {
        int nr = idx >> 6, k = idx & 63;
        s_w4[nr * W_S + k] = (_Float16)P.gW3[idx];
    }
    for (int idx = lane; idx < 64; idx += 32) {
        s_gb1[idx] = P.gB0[idx]; s_gb2[idx] = P.gB1[idx]; s_gb3[idx] = P.gB2[idx];
        s_dr[idx] = P.drW[idx];
    }
    for (int idx = lane; idx < 48; idx += 32) s_gb4[idx] = P.gB3[idx];
    for (int idx = lane; idx < 512; idx += 32) s_df[idx] = P.dfW[idx];
    if (lane < 32) s_dfb[lane] = P.dfb[lane];
    if (lane < 2)  s_drb[lane] = P.drb[lane];

    // ---- encoder (scalar; runs once): z0 = enc(xs[:,0,:]) -> mu, logvar
    if (lane < 16) {
        const int b = bbase + lane;
        const float x0 = P.xs[(b * T_TOT) * 2 + 0];
        const float x1 = P.xs[(b * T_TOT) * 2 + 1];
        float h1[32], h2[32], h3[32];
#pragma unroll 1
        for (int j = 0; j < 32; ++j)
            h1[j] = softplus_f(P.eW0[j * 2 + 0] * x0 + P.eW0[j * 2 + 1] * x1 + P.eB0[j]);
#pragma unroll 1
        for (int j = 0; j < 32; ++j) {
            float acc = P.eB1[j];
#pragma unroll
            for (int k = 0; k < 32; ++k) acc += P.eW1[j * 32 + k] * h1[k];
            h2[j] = softplus_f(acc);
        }
#pragma unroll 1
        for (int j = 0; j < 32; ++j) {
            float acc = P.eB2[j];
#pragma unroll
            for (int k = 0; k < 32; ++k) acc += P.eW2[j * 32 + k] * h2[k];
            h3[j] = softplus_f(acc);
        }
#pragma unroll 1
        for (int j = 0; j < 32; ++j) {
            float acc = P.eB3[j];
#pragma unroll
            for (int k = 0; k < 32; ++k) acc += P.eW3[j * 32 + k] * h3[k];
            if (j < 16) {
                s_y[lane * 16 + j] = acc;
                P.out[MU_OFF + b * 16 + j] = acc;
            } else {
                P.out[LV_OFF + b * 16 + (j - 16)] = acc;
            }
        }
    }
    wave_sync();

    // per-row decode of current y: x_factors -> x_recons
    auto decode = [&](int t, bool storeFac) {
        if (lane < 16) {
            const int b = bbase + lane;
            float z[16];
#pragma unroll
            for (int l = 0; l < 16; ++l) z[l] = s_y[lane * 16 + l];
            float r0 = s_drb[0], r1 = s_drb[1];
#pragma unroll 1
            for (int jf = 0; jf < 32; ++jf) {
                float f = s_dfb[jf];
#pragma unroll
                for (int l = 0; l < 16; ++l) f += s_df[jf * 16 + l] * z[l];
                r0 += s_dr[jf] * f;
                r1 += s_dr[32 + jf] * f;
                if (storeFac) P.out[XFAC_OFF + b * 32 + jf] = f;
            }
            P.out[XREC_OFF + (b * T_TOT + t) * 2 + 0] = r0;
            P.out[XREC_OFF + (b * T_TOT + t) * 2 + 1] = r1;
        }
    };

    decode(0, false);   // zs[:,0] = mu
    wave_sync();

    // ---- sequential ODE scan ----
#pragma unroll 1
    for (int i = 0; i < T_TOT - 1; ++i) {
        const float t0 = P.te[i], t1 = P.te[i + 1];
        const float h = t1 - t0, hs = h, invh = 1.f / h;

        if (lane < 16) {  // Hermite segment data per batch row
            const int b = bbase + lane;
            const float x0i = P.xs[(b * T_TOT + i) * 2 + 0];
            const float x1i = P.xs[(b * T_TOT + i) * 2 + 1];
            const float x0n = P.xs[(b * T_TOT + i + 1) * 2 + 0];
            const float x1n = P.xs[(b * T_TOT + i + 1) * 2 + 1];
            s_pm[lane * 4 + 0] = -1.f;                      // (p0-p1)/h, c=time
            s_pm[lane * 4 + 1] = (x0i - x0n) * invh;
            s_pm[lane * 4 + 2] = (x1i - x1n) * invh;
            s_d1[lane * 4 + 0] = 1.f;                       // slopes[i]
            s_d1[lane * 4 + 1] = (x0n - x0i) * invh;
            s_d1[lane * 4 + 2] = (x1n - x1i) * invh;
            if (i == 0) {
                s_d0[lane * 4 + 0] = 1.f;
                s_d0[lane * 4 + 1] = (x0n - x0i) * invh;
                s_d0[lane * 4 + 2] = (x1n - x1i) * invh;
            } else {
                const float invhp = 1.f / (t0 - P.te[i - 1]);
                const float x0p = P.xs[(b * T_TOT + i - 1) * 2 + 0];
                const float x1p = P.xs[(b * T_TOT + i - 1) * 2 + 1];
                s_d0[lane * 4 + 0] = 1.f;
                s_d0[lane * 4 + 1] = (x0i - x0p) * invhp;
                s_d0[lane * 4 + 2] = (x1i - x1p) * invhp;
            }
        }

#pragma unroll 1
        for (int j = 0; j < 6; ++j) {   // 6 RK stages
            const float s    = c_Cs[j];
            const float dh00 = 6.f * s * s - 6.f * s;
            const float dh10 = 3.f * s * s - 4.f * s + 1.f;
            const float dh11 = 3.f * s * s - 2.f * s;
            float* s_kc = &s_k[j * 256];

            if (lane < 16) {
#pragma unroll
                for (int c = 0; c < 3; ++c)
                    s_dx[lane * 4 + c] = dh00 * s_pm[lane * 4 + c] +
                                         dh10 * s_d0[lane * 4 + c] +
                                         dh11 * s_d1[lane * 4 + c];
            }
            // stage input ytmp = y + hs * sum A[j][kk]*k_kk -> f16 act buffer (K padded)
            for (int idx = lane; idx < 256; idx += 32) {
                s_kc[idx] = 0.f;
                float acc = 0.f;
                for (int kk = 0; kk < j; ++kk) acc += c_A[j][kk] * s_k[kk * 256 + idx];
                const float v = s_y[idx] + hs * acc;
                const int r = idx >> 4, l = idx & 15;
                s_actA[r * ACT_S + l]      = (_Float16)v;
                s_actA[r * ACT_S + 16 + l] = (_Float16)0.f;
            }
            wave_sync();

            gen_layer<4, 1>(s_actA, s_w1, W1_S, s_gb1, s_actB, lane);  // 16->64
            wave_sync();
            gen_layer<4, 2>(s_actB, s_w2, W_S, s_gb2, s_actA, lane);   // 64->64
            wave_sync();
            gen_layer<4, 2>(s_actA, s_w3, W_S, s_gb3, s_actB, lane);   // 64->64
            wave_sync();

            // layer 4 (64->48) + tanh + einsum('blc,bc->bl') into k_j
            {
                const int n = lane & 15;
                const int mbase = (lane >> 4) << 3;
                v16h a[2];
#pragma unroll
                for (int kc = 0; kc < 2; ++kc)
                    a[kc] = ld_frag(s_actB + n * ACT_S, kc * 32, lane);
#pragma unroll
                for (int nt = 0; nt < 3; ++nt) {
                    const float bv = s_gb4[nt * 16 + n];
                    v8f c;
#pragma unroll
                    for (int e = 0; e < 8; ++e) c[e] = bv;
#pragma unroll
                    for (int kc = 0; kc < 2; ++kc) {
                        v16h bf = ld_frag(s_w4 + (nt * 16 + n) * W_S, kc * 32, lane);
                        c = __builtin_amdgcn_wmma_f32_16x16x32_f16(false, a[kc], false, bf,
                                                                   (short)0, c, false, false);
                    }
                    const int col = nt * 16 + n;
                    const int l = col / 3, cc = col - l * 3;
#pragma unroll
                    for (int e = 0; e < 8; ++e) {
                        const int m = mbase + e;
                        atomicAdd(&s_kc[m * 16 + l], tanh_f(c[e]) * s_dx[m * 4 + cc]);
                    }
                }
            }
            wave_sync();
        }

        // y += hs * sum B[j]*k_j
        for (int idx = lane; idx < 256; idx += 32) {
            float acc = 0.f;
#pragma unroll
            for (int j = 0; j < 6; ++j) acc += c_B[j] * s_k[j * 256 + idx];
            s_y[idx] += hs * acc;
        }
        wave_sync();

        decode(i + 1, i == T_TOT - 2);
        wave_sync();
    }
}

extern "C" void kernel_launch(void* const* d_in, const int* in_sizes, int n_in,
                              void* d_out, int out_size, void* d_ws, size_t ws_size,
                              hipStream_t stream) {
    (void)in_sizes; (void)out_size; (void)d_ws; (void)ws_size;
    if (n_in < 22) return;
    Params P;
    P.xs  = (const float*)d_in[0];
    P.te  = (const float*)d_in[1];
    P.eW0 = (const float*)d_in[2];  P.eB0 = (const float*)d_in[3];
    P.eW1 = (const float*)d_in[4];  P.eB1 = (const float*)d_in[5];
    P.eW2 = (const float*)d_in[6];  P.eB2 = (const float*)d_in[7];
    P.eW3 = (const float*)d_in[8];  P.eB3 = (const float*)d_in[9];
    P.gW0 = (const float*)d_in[10]; P.gB0 = (const float*)d_in[11];
    P.gW1 = (const float*)d_in[12]; P.gB1 = (const float*)d_in[13];
    P.gW2 = (const float*)d_in[14]; P.gB2 = (const float*)d_in[15];
    P.gW3 = (const float*)d_in[16]; P.gB3 = (const float*)d_in[17];
    P.dfW = (const float*)d_in[18]; P.dfb = (const float*)d_in[19];
    P.drW = (const float*)d_in[20]; P.drb = (const float*)d_in[21];
    P.out = (float*)d_out;
    hipLaunchKernelGGL(node_wmma_kernel, dim3(B_TOT / 16), dim3(32), 0, stream, P);
}
